// MemoryEfficientPatterns_13932873908470
// MI455X (gfx1250) — matrix operand order repaired and checked by reading, not compiled, not run on hardware
//
#include <hip/hip_runtime.h>
#include <cstdint>

// ---------------------------------------------------------------------------
// Problem constants (match reference)
// ---------------------------------------------------------------------------
#define BATCH 2
#define SEQ   2048
#define HDIM  2048           // hidden size H
#define NH    16             // query heads
#define NKV   4              // kv heads
#define HD    128            // head dim
#define QDIM  (NH * HD)      // 2048
#define KVDIM (NKV * HD)     // 512
#define ROWS  (BATCH * SEQ)  // 4096 token rows

typedef __attribute__((ext_vector_type(16))) __bf16 v16bf;
typedef __attribute__((ext_vector_type(8)))  __bf16 v8bf;
typedef __attribute__((ext_vector_type(8)))  float  v8f;
typedef __attribute__((ext_vector_type(4)))  unsigned int u32x4;
typedef __attribute__((ext_vector_type(8)))  unsigned int u32x8;

// ---------------------------------------------------------------------------
// WMMA helper: D = A(16x32 bf16) * B(32x16 bf16) + C(16x16 f32)
// ---------------------------------------------------------------------------
__device__ inline v8f wmma_bf16(v16bf a, v16bf b, v8f c) {
    return __builtin_amdgcn_wmma_f32_16x16x32_bf16(
        /*neg_a=*/false, a, /*neg_b=*/false, b,
        /*c_mod=*/(short)0, c, /*reuse_a=*/false, /*reuse_b=*/false);
}

// A fragment (16x32, MxK), row-major source with leading dim ld.
// Layout: lanes 0-15 -> M=lane, elems 0-15 = K 0-15;
//         lanes 16-31 -> M=lane-16, elems 0-15 = K 16-31 (contiguous, matches B).
__device__ inline v16bf load_a16x32(const __bf16* base, int ld, int m0, int k0, int lane) {
    int row = m0 + (lane & 15);
    int hlf = lane >> 4;
    const __bf16* p = base + (size_t)row * ld + k0 + hlf * 16;
    v8bf lo = *(const v8bf*)p;        // K = k0 + hlf*16 .. +7
    v8bf hi = *(const v8bf*)(p + 8);  // K = k0 + hlf*16 + 8 .. +15
    return __builtin_shufflevector(lo, hi, 0,1,2,3,4,5,6,7,8,9,10,11,12,13,14,15);
}

// B fragment (32x16, KxN) from a row-major (N,K) matrix (i.e. computes A*W^T).
// Layout used here (same k order as A): lane L holds column N = L%16, elems i = K = (L/16)*16 + i  (contiguous)
__device__ inline v16bf load_b32x16_wt(const __bf16* base, int ld, int n0, int k0, int lane) {
    int n = n0 + (lane & 15);
    int k = k0 + (lane >> 4) * 16;
    return *(const v16bf*)(base + (size_t)n * ld + k);
}

// ---------------------------------------------------------------------------
// Tensor Data Mover: DMA a 2D tile (128 rows x 64 bf16) from global into LDS.
// Builds D# group0 (128b) + group1 (256b) per CDNA5 ISA
// and issues TENSOR_LOAD_TO_LDS (<=2D form: groups 2/3 omitted).
// Completion tracked with TENSORcnt (in-order per wave).
// ---------------------------------------------------------------------------
__device__ inline void tdm_load_tile_128x64(const __bf16* lds_dst, const __bf16* gsrc,
                                            unsigned row_stride_elems) {
    unsigned lds_off = (unsigned)(uintptr_t)lds_dst;  // low 32 bits = LDS byte offset
    unsigned long long ga = (unsigned long long)(uintptr_t)gsrc;
    u32x4 g0;
    g0[0] = 1u;                                       // count=1, user mode
    g0[1] = lds_off;                                  // lds_addr
    g0[2] = (unsigned)ga;                             // global_addr[31:0]
    g0[3] = (unsigned)((ga >> 32) & 0x1ffffffu)       // global_addr[56:32]
          | (2u << 30);                               // type = 2 ("image")
    u32x8 g1;
    g1[0] = 1u << 16;                                 // data_size = 1 -> 2 bytes
    g1[1] = 64u << 16;                                // tensor_dim0[15:0] = 64
    g1[2] = 128u << 16;                               // tensor_dim0 hi=0 | tensor_dim1 = 128
    g1[3] = 64u << 16;                                // tensor_dim1 hi=0 | tile_dim0 = 64
    g1[4] = 128u;                                     // tile_dim1 = 128, tile_dim2 = 0
    g1[5] = row_stride_elems;                         // tensor_dim0_stride[31:0] (elems)
    g1[6] = 0u;                                       // stride0 hi | stride1 lo
    g1[7] = 0u;
    asm volatile("tensor_load_to_lds %0, %1" :: "s"(g0), "s"(g1) : "memory");
}

// ---------------------------------------------------------------------------
// fp32 -> bf16 conversion
// ---------------------------------------------------------------------------
__global__ __launch_bounds__(256)
void f32_to_bf16_kernel(const float* __restrict__ in, __bf16* __restrict__ out, int n) {
    int i = blockIdx.x * 256 + threadIdx.x;
    if (i < n) out[i] = (__bf16)in[i];
}

// ---------------------------------------------------------------------------
// GEMM: out[M,N] = A[M,K](bf16) * W[N,K](bf16)^T + bias[N]
// 256 threads (8 waves: 4 M x 2 N), block tile 128x128, K step 64.
// TDM double-buffered LDS staging: wave 0 issues the next K-slice's two tile
// DMAs, waits tensorcnt<=2 (in-order => current slice complete), barrier, then
// all waves run 16 WMMAs out of LDS.
// ---------------------------------------------------------------------------
template <bool BF16_OUT>
__global__ __launch_bounds__(256)
void gemm_bias_tdm_kernel(const __bf16* __restrict__ A, const __bf16* __restrict__ W,
                          const float* __restrict__ bias, void* __restrict__ out,
                          int M, int N, int K) {
    __shared__ __bf16 As[2][128][64];   // [buf][m][k]  16 KB each
    __shared__ __bf16 Bs[2][128][64];   // [buf][n][k]
    const int lane  = threadIdx.x & 31;
    const int wave  = threadIdx.x >> 5;
    const int waveM = wave & 3;
    const int waveN = wave >> 2;
    const int mblk  = blockIdx.y * 128;
    const int nblk  = blockIdx.x * 128;
    const int hlf   = lane >> 4;

    v8f acc[2][4];
#pragma unroll
    for (int i = 0; i < 2; ++i)
#pragma unroll
        for (int j = 0; j < 4; ++j)
            acc[i][j] = v8f{0.f,0.f,0.f,0.f,0.f,0.f,0.f,0.f};

    const int nkt = K >> 6;  // K / 64
    if (threadIdx.x < 32) {  // wave 0 drives the DMA pipeline
        tdm_load_tile_128x64(&As[0][0][0], A + (size_t)mblk * K, (unsigned)K);
        tdm_load_tile_128x64(&Bs[0][0][0], W + (size_t)nblk * K, (unsigned)K);
    }

    for (int kt = 0; kt < nkt; ++kt) {
        const int buf = kt & 1;
        if (threadIdx.x < 32) {
            if (kt + 1 < nkt) {
                tdm_load_tile_128x64(&As[buf ^ 1][0][0],
                                     A + (size_t)mblk * K + (kt + 1) * 64, (unsigned)K);
                tdm_load_tile_128x64(&Bs[buf ^ 1][0][0],
                                     W + (size_t)nblk * K + (kt + 1) * 64, (unsigned)K);
                __builtin_amdgcn_s_wait_tensorcnt(2);  // current slice's 2 DMAs done
            } else {
                __builtin_amdgcn_s_wait_tensorcnt(0);
            }
        }
        __syncthreads();  // staged tile visible to all waves

        const __bf16* Abase = &As[buf][waveM * 32][0];
        const __bf16* Bbase = &Bs[buf][waveN * 64][0];
#pragma unroll
        for (int ks = 0; ks < 2; ++ks) {
            v16bf a0 = load_a16x32(Abase, 64, 0,  ks * 32, lane);
            v16bf a1 = load_a16x32(Abase, 64, 16, ks * 32, lane);
#pragma unroll
            for (int j = 0; j < 4; ++j) {
                v16bf b = load_b32x16_wt(Bbase, 64, j * 16, ks * 32, lane);
                acc[0][j] = wmma_bf16(a0, b, acc[0][j]);
                acc[1][j] = wmma_bf16(a1, b, acc[1][j]);
            }
        }
        __syncthreads();  // all reads done before next iteration's DMA overwrites
    }

    const int m0 = mblk + waveM * 32;
    const int n0 = nblk + waveN * 64;
#pragma unroll
    for (int j = 0; j < 4; ++j) {
        int col = n0 + j * 16 + (lane & 15);
        float bv = bias[col];
#pragma unroll
        for (int i = 0; i < 2; ++i)
#pragma unroll
            for (int r = 0; r < 8; ++r) {
                int row = m0 + i * 16 + hlf * 8 + r;
                float v = acc[i][j][r] + bv;
                if (BF16_OUT)
                    ((__bf16*)out)[(size_t)row * N + col] = (__bf16)v;
                else
                    ((float*)out)[(size_t)row * N + col] = v;
            }
    }
}

// ---------------------------------------------------------------------------
// In-place RoPE on bf16 (B, S, nheads*HD); one thread per (row, head, pair j)
// ---------------------------------------------------------------------------
__global__ __launch_bounds__(256)
void rope_kernel(__bf16* __restrict__ x, int nheads, int total) {
    int idx = blockIdx.x * 256 + threadIdx.x;
    if (idx >= total) return;
    int j  = idx & 63;          // pair index 0..63
    int t  = idx >> 6;          // (b*S+s)*nheads + h
    int hh = t % nheads;
    int bs = t / nheads;        // b*S + s
    int s  = bs & (SEQ - 1);
    float inv_freq = __powf(10000.0f, -(float)(2 * j) * (1.0f / (float)HD));
    float ang = (float)s * inv_freq;
    float sn, cs;
    __sincosf(ang, &sn, &cs);
    __bf16* p = x + ((size_t)bs * nheads + hh) * HD;
    float x1 = (float)p[j];
    float x2 = (float)p[j + 64];
    p[j]      = (__bf16)(x1 * cs - x2 * sn);
    p[j + 64] = (__bf16)(x2 * cs + x1 * sn);
}

// ---------------------------------------------------------------------------
// Flash attention (non-causal, online softmax), GQA 4:1.
// grid = (S/64, NH, B), block = 128 (4 waves). Each wave owns 16 q rows.
// Per 32-key chunk: 8 WMMAs for scores + 8 WMMAs for P*V.
// ---------------------------------------------------------------------------
__global__ __launch_bounds__(128)
void flash_attn_kernel(const __bf16* __restrict__ Q, const __bf16* __restrict__ Kb,
                       const __bf16* __restrict__ Vb, __bf16* __restrict__ O) {
    __shared__ __bf16 Vt[HD][32];      // transposed V chunk: [hd][key] (8 KB)
    __shared__ __bf16 Pt[4][16][32];   // per-wave probability tile (4 KB)

    const int lane = threadIdx.x & 31;
    const int wave = threadIdx.x >> 5;
    const int hlf  = lane >> 4;
    const int h    = blockIdx.y;
    const int b    = blockIdx.z;
    const int kvh  = h >> 2;           // repeat_interleave -> q head h uses kv head h/4
    const int qbase = blockIdx.x * 64 + wave * 16;

    const __bf16* Qbase = Q + ((size_t)(b * SEQ + qbase)) * QDIM + h * HD;
    const __bf16* Kbase = Kb + ((size_t)(b * SEQ)) * KVDIM + kvh * HD;
    const __bf16* Vbase = Vb + ((size_t)(b * SEQ)) * KVDIM + kvh * HD;

    // Preload Q fragments for this wave's 16 rows (4 hd-slices of 32)
    v16bf aq[4];
#pragma unroll
    for (int sl = 0; sl < 4; ++sl) aq[sl] = load_a16x32(Qbase, QDIM, 0, sl * 32, lane);

    v8f acc[8];
#pragma unroll
    for (int t = 0; t < 8; ++t) acc[t] = v8f{0.f,0.f,0.f,0.f,0.f,0.f,0.f,0.f};
    float mrow[8], lrow[8];
#pragma unroll
    for (int r = 0; r < 8; ++r) { mrow[r] = -3.0e38f; lrow[r] = 0.0f; }

    const float scale = 0.08838834764831845f;  // 1/sqrt(128)

    for (int kc = 0; kc < SEQ; kc += 32) {
        // --- cooperatively stage transposed V chunk (32 keys x 128 hd) ---
        {
            int tid = threadIdx.x;
            int key = tid >> 2;            // 0..31
            int d0  = (tid & 3) * 32;      // 0,32,64,96
            const __bf16* vp = Vbase + (size_t)(kc + key) * KVDIM + d0;
            v8bf tmp[4];
#pragma unroll
            for (int jj = 0; jj < 4; ++jj) tmp[jj] = *(const v8bf*)(vp + jj * 8);
#pragma unroll
            for (int j = 0; j < 32; ++j) Vt[d0 + j][key] = tmp[j >> 3][j & 7];
        }
        __syncthreads();

        // --- scores: two 16-key tiles, K-dim = 128 hd in 4 slices ---
        v8f s0 = v8f{0.f,0.f,0.f,0.f,0.f,0.f,0.f,0.f};
        v8f s1 = v8f{0.f,0.f,0.f,0.f,0.f,0.f,0.f,0.f};
#pragma unroll
        for (int sl = 0; sl < 4; ++sl) {
            v16bf bk0 = load_b32x16_wt(Kbase, KVDIM, kc,      sl * 32, lane);
            v16bf bk1 = load_b32x16_wt(Kbase, KVDIM, kc + 16, sl * 32, lane);
            s0 = wmma_bf16(aq[sl], bk0, s0);
            s1 = wmma_bf16(aq[sl], bk1, s1);
        }

        // --- online softmax. C layout: lane = key column within tile, elem r = row hlf*8+r.
        // Row-wise reductions happen across the 16 lanes of each half-wave. ---
        float alpha[8];
#pragma unroll
        for (int r = 0; r < 8; ++r) {
            float x0 = s0[r] * scale;
            float x1 = s1[r] * scale;
            float mx = fmaxf(x0, x1);
#pragma unroll
            for (int msk = 1; msk <= 8; msk <<= 1)
                mx = fmaxf(mx, __shfl_xor(mx, msk, 32));
            float nm = fmaxf(mrow[r], mx);
            alpha[r] = __expf(mrow[r] - nm);
            mrow[r] = nm;
            float p0 = __expf(x0 - nm);
            float p1 = __expf(x1 - nm);
            float rs = p0 + p1;
#pragma unroll
            for (int msk = 1; msk <= 8; msk <<= 1)
                rs += __shfl_xor(rs, msk, 32);
            lrow[r] = lrow[r] * alpha[r] + rs;
            int prow = hlf * 8 + r;
            Pt[wave][prow][lane & 15]        = (__bf16)p0;
            Pt[wave][prow][16 + (lane & 15)] = (__bf16)p1;
        }
#pragma unroll
        for (int t = 0; t < 8; ++t)
#pragma unroll
            for (int r = 0; r < 8; ++r)
                acc[t][r] = acc[t][r] * alpha[r];

        // --- P (C layout) -> A layout via per-wave LDS tile (DS ops in-order per wave)
        v16bf ap = load_a16x32(&Pt[wave][0][0], 32, 0, 0, lane);

        // --- P * V: 8 hd tiles; B frag from transposed V in LDS: lane = hd col,
        // elems = 16 contiguous keys ---
#pragma unroll
        for (int t = 0; t < 8; ++t) {
            v16bf bv = *(const v16bf*)&Vt[t * 16 + (lane & 15)][hlf * 16];
            acc[t] = wmma_bf16(ap, bv, acc[t]);
        }
        __syncthreads();
    }

    // --- normalize and write O (b, s, h*HD + d) as bf16 ---
#pragma unroll
    for (int r = 0; r < 8; ++r) {
        int qrow = qbase + hlf * 8 + r;
        float inv = 1.0f / (lrow[r] + 1e-10f);
        __bf16* op = O + ((size_t)(b * SEQ + qrow)) * QDIM + h * HD;
#pragma unroll
        for (int t = 0; t < 8; ++t)
            op[t * 16 + (lane & 15)] = (__bf16)(acc[t][r] * inv);
    }
}

// ---------------------------------------------------------------------------
// Host launcher
// ---------------------------------------------------------------------------
extern "C" void kernel_launch(void* const* d_in, const int* in_sizes, int n_in,
                              void* d_out, int out_size, void* d_ws, size_t ws_size,
                              hipStream_t stream) {
    const float* hidden = (const float*)d_in[0];
    const float* q_w = (const float*)d_in[1];
    const float* q_b = (const float*)d_in[2];
    const float* k_w = (const float*)d_in[3];
    const float* k_b = (const float*)d_in[4];
    const float* v_w = (const float*)d_in[5];
    const float* v_b = (const float*)d_in[6];
    const float* o_w = (const float*)d_in[7];
    const float* o_b = (const float*)d_in[8];
    float* out = (float*)d_out;

    char* ws = (char*)d_ws;
    // workspace layout (bytes)
    __bf16* hb  = (__bf16*)(ws + 0);          // 4096x2048  hidden bf16
    __bf16* qwb = (__bf16*)(ws + 16777216);   // 2048x2048
    __bf16* kwb = (__bf16*)(ws + 25165824);   // 512x2048
    __bf16* vwb = (__bf16*)(ws + 27262976);   // 512x2048
    __bf16* owb = (__bf16*)(ws + 29360128);   // 2048x2048
    __bf16* qp  = (__bf16*)(ws + 37748736);   // 4096x2048  Q (rope applied in place)
    __bf16* kp  = (__bf16*)(ws + 54525952);   // 4096x512   K (rope applied in place)
    __bf16* vr  = (__bf16*)(ws + 58720256);   // 4096x512   V
    __bf16* att = (__bf16*)(ws + 62914560);   // 4096x2048  attention output

    // 1) fp32 -> bf16 conversions
    {
        int n;
        n = ROWS * HDIM;  f32_to_bf16_kernel<<<n / 256, 256, 0, stream>>>(hidden, hb, n);
        n = QDIM * HDIM;  f32_to_bf16_kernel<<<n / 256, 256, 0, stream>>>(q_w, qwb, n);
        n = KVDIM * HDIM; f32_to_bf16_kernel<<<n / 256, 256, 0, stream>>>(k_w, kwb, n);
        n = KVDIM * HDIM; f32_to_bf16_kernel<<<n / 256, 256, 0, stream>>>(v_w, vwb, n);
        n = HDIM * QDIM;  f32_to_bf16_kernel<<<n / 256, 256, 0, stream>>>(o_w, owb, n);
    }

    // 2) QKV projections (bf16 out), TDM-staged WMMA GEMM
    gemm_bias_tdm_kernel<true><<<dim3(QDIM / 128, ROWS / 128), 256, 0, stream>>>(
        hb, qwb, q_b, (void*)qp, ROWS, QDIM, HDIM);
    gemm_bias_tdm_kernel<true><<<dim3(KVDIM / 128, ROWS / 128), 256, 0, stream>>>(
        hb, kwb, k_b, (void*)kp, ROWS, KVDIM, HDIM);
    gemm_bias_tdm_kernel<true><<<dim3(KVDIM / 128, ROWS / 128), 256, 0, stream>>>(
        hb, vwb, v_b, (void*)vr, ROWS, KVDIM, HDIM);

    // 3) RoPE in place on Q and K
    {
        int totq = ROWS * NH * 64;
        rope_kernel<<<totq / 256, 256, 0, stream>>>(qp, NH, totq);
        int totk = ROWS * NKV * 64;
        rope_kernel<<<totk / 256, 256, 0, stream>>>(kp, NKV, totk);
    }

    // 4) attention
    flash_attn_kernel<<<dim3(SEQ / 64, NH, BATCH), 128, 0, stream>>>(qp, kp, vr, att);

    // 5) output projection (f32 out with bias)
    gemm_bias_tdm_kernel<false><<<dim3(HDIM / 128, ROWS / 128), 256, 0, stream>>>(
        att, owb, o_b, (void*)out, ROWS, HDIM, QDIM);
}
